// GroupedExperts_64372969833101
// MI455X (gfx1250) — matrix operand order repaired and checked by reading, not compiled, not run on hardware
//
#include <hip/hip_runtime.h>
#include <hip/hip_bf16.h>

#define E_    8
#define T_    2048
#define DIN_  2048
#define DOUT_ 1408
#define LSTR  40  // padded LDS row stride in f16 elements (80B = 20 banks)

typedef __attribute__((ext_vector_type(16))) _Float16     v16h;
typedef __attribute__((ext_vector_type(8)))  _Float16     v8h;
typedef __attribute__((ext_vector_type(8)))  float        v8f;

// ---------------------------------------------------------------------------
// CDNA5 async global->LDS copy (16B per lane), tracked by ASYNCcnt.
//   global_load_async_to_lds_b128 vdst(lds byte addr), vaddr(u32 byte off),
//                                 saddr(64-bit base)
// ---------------------------------------------------------------------------
__device__ __forceinline__ void async_ld16(unsigned lds_byte, const void* sbase,
                                           unsigned gbyte) {
  asm volatile("global_load_async_to_lds_b128 %0, %1, %2"
               :
               : "v"(lds_byte), "v"(gbyte), "s"(sbase)
               : "memory");
}
__device__ __forceinline__ void wait_async0() {
  asm volatile("s_wait_asynccnt 0x0" ::: "memory");
}
__device__ __forceinline__ void wait_async4() {
  asm volatile("s_wait_asynccnt 0x4" ::: "memory");
}
__device__ __forceinline__ void wait_async6() {
  asm volatile("s_wait_asynccnt 0x6" ::: "memory");
}

__device__ __forceinline__ unsigned lds_base_u32(const void* p) {
  // generic pointers to LDS carry the workgroup-relative offset in addr[31:0]
  return (unsigned)(unsigned long long)p;
}

// ---------------------------------------------------------------------------
// f32 -> f16 elementwise convert (vectorized x4)
// ---------------------------------------------------------------------------
__global__ __launch_bounds__(256) void k_cvt_f16(const float* __restrict__ in,
                                                 _Float16* __restrict__ out,
                                                 long long n) {
  long long i = ((long long)blockIdx.x * blockDim.x + threadIdx.x) * 4;
  if (i + 3 < n) {
    float4 v = *(const float4*)(in + i);
    out[i + 0] = (_Float16)v.x;
    out[i + 1] = (_Float16)v.y;
    out[i + 2] = (_Float16)v.z;
    out[i + 3] = (_Float16)v.w;
  } else {
    for (long long j = i; j < n && j >= 0; ++j) out[j] = (_Float16)in[j];
  }
}

// ---------------------------------------------------------------------------
// Tiled transpose + f32->f16 convert: in [e][R][C] f32 -> out [e][C][R] f16
// ---------------------------------------------------------------------------
__global__ __launch_bounds__(256) void k_transpose_cvt(
    const float* __restrict__ in, _Float16* __restrict__ out, int R, int C) {
  __shared__ float tile[32][33];
  const int e = blockIdx.z;
  const float* src = in + (size_t)e * R * C;
  _Float16* dst = out + (size_t)e * R * C;
  const int r0 = blockIdx.x * 32;
  const int c0 = blockIdx.y * 32;
  const int tx = threadIdx.x;
  const int ty = threadIdx.y;
#pragma unroll
  for (int i = 0; i < 4; ++i) {
    const int r = r0 + ty + i * 8;
    tile[ty + i * 8][tx] = src[(size_t)r * C + (c0 + tx)];
  }
  __syncthreads();
#pragma unroll
  for (int i = 0; i < 4; ++i) {
    const int c = c0 + ty + i * 8;
    dst[(size_t)c * R + (r0 + tx)] = (_Float16)tile[tx][ty + i * 8];
  }
}

// ---------------------------------------------------------------------------
// WMMA fragment helpers (wave32, ISA 7.12.2 layouts), padded LDS stride LSTR
// ---------------------------------------------------------------------------
__device__ __forceinline__ v16h load_a_frag(const _Float16* lds, int row,
                                            int lane) {
  const int kb = (lane >> 4) * 8;
  const _Float16* p = lds + row * LSTR + kb;
  v8h lo = *(const v8h*)(p);
  v8h hi = *(const v8h*)(p + 16);
  v16h a;
#pragma unroll
  for (int i = 0; i < 8; ++i) {
    a[i] = lo[i];
    a[i + 8] = hi[i];
  }
  return a;
}

__device__ __forceinline__ v16h load_b_frag(const _Float16* lds, int col,
                                            int lane) {
  const int kb = (lane >> 4) * 8;
  const _Float16* p = lds + col * LSTR + kb;
  v8h lo = *(const v8h*)(p);
  v8h hi = *(const v8h*)(p + 16);
  v16h b;
#pragma unroll
  for (int i = 0; i < 8; ++i) {
    b[i] = lo[i];
    b[i + 8] = hi[i];
  }
  return b;
}

// ---------------------------------------------------------------------------
// GEMM1 (fused SwiGLU): h = silu(x@Wg) * (x@Wd), f16 out.
// Block tile 128(M) x 128(N), K-step 32, double-buffered async LDS staging.
// 8 waves, each 64(M) x 32(N) -> 16 WMMA per wave per K-step.
// ---------------------------------------------------------------------------
__global__ __launch_bounds__(256) void k_gemm1_swiglu(
    const _Float16* __restrict__ xb, const _Float16* __restrict__ gtb,
    const _Float16* __restrict__ dtb, _Float16* __restrict__ hb) {
  __shared__ __align__(16) _Float16 lds_a[2][128 * LSTR];
  __shared__ __align__(16) _Float16 lds_g[2][128 * LSTR];
  __shared__ __align__(16) _Float16 lds_d[2][128 * LSTR];

  const int e = blockIdx.z;
  const int m0 = blockIdx.x * 128;
  const int n0 = blockIdx.y * 128;
  const _Float16* Ae = xb + (size_t)e * T_ * DIN_;
  const _Float16* Ge = gtb + (size_t)e * DOUT_ * DIN_ + (size_t)n0 * DIN_;
  const _Float16* De = dtb + (size_t)e * DOUT_ * DIN_ + (size_t)n0 * DIN_;

  const int tid = threadIdx.x;
  const int lane = tid & 31;
  const int w = tid >> 5;
  const int wm = (w & 1) * 64;   // 2 waves along M
  const int wn = (w >> 1) * 32;  // 4 waves along N
  const int lmod = lane & 15;
  const int lhi = lane >> 4;

  // staging chunk coords: 512 x 16B chunks per 128x32 tile; thread does
  // chunks {tid, tid+256}.
  const int row0 = tid >> 2;
  const int row1 = (tid + 256) >> 2;
  const int col = (tid & 3) * 8;

  unsigned la[2], lg[2], ld[2];
#pragma unroll
  for (int b = 0; b < 2; ++b) {
    la[b] = lds_base_u32(&lds_a[b][0]);
    lg[b] = lds_base_u32(&lds_g[b][0]);
    ld[b] = lds_base_u32(&lds_d[b][0]);
  }

  auto stage = [&](int buf, int k0) {
    const unsigned o0 = (unsigned)(row0 * LSTR + col) * 2u;
    const unsigned o1 = (unsigned)(row1 * LSTR + col) * 2u;
    const unsigned ga0 = (unsigned)((m0 + row0) * DIN_ + k0 + col) * 2u;
    const unsigned ga1 = (unsigned)((m0 + row1) * DIN_ + k0 + col) * 2u;
    const unsigned gb0 = (unsigned)(row0 * DIN_ + k0 + col) * 2u;
    const unsigned gb1 = (unsigned)(row1 * DIN_ + k0 + col) * 2u;
    async_ld16(la[buf] + o0, Ae, ga0);
    async_ld16(la[buf] + o1, Ae, ga1);
    async_ld16(lg[buf] + o0, Ge, gb0);
    async_ld16(lg[buf] + o1, Ge, gb1);
    async_ld16(ld[buf] + o0, De, gb0);
    async_ld16(ld[buf] + o1, De, gb1);
  };

  v8f accg[4][2] = {};
  v8f accd[4][2] = {};

  stage(0, 0);
  const int KT = DIN_ / 32;
  for (int kt = 0; kt < KT; ++kt) {
    const int cur = kt & 1;
    if (kt + 1 < KT) {
      stage(cur ^ 1, (kt + 1) * 32);
      wait_async6();  // previous 6 done; 6 new still in flight
    } else {
      wait_async0();
    }
    __syncthreads();

    const _Float16* A = &lds_a[cur][0];
    const _Float16* G = &lds_g[cur][0];
    const _Float16* D = &lds_d[cur][0];

    v16h af[4], bgf[2], bdf[2];
#pragma unroll
    for (int jm = 0; jm < 4; ++jm)
      af[jm] = load_a_frag(A, wm + jm * 16 + lmod, lane);
#pragma unroll
    for (int jn = 0; jn < 2; ++jn) {
      bgf[jn] = load_b_frag(G, wn + jn * 16 + lmod, lane);
      bdf[jn] = load_b_frag(D, wn + jn * 16 + lmod, lane);
    }
#pragma unroll
    for (int jm = 0; jm < 4; ++jm)
#pragma unroll
      for (int jn = 0; jn < 2; ++jn) {
        accg[jm][jn] = __builtin_amdgcn_wmma_f32_16x16x32_f16(
            false, af[jm], false, bgf[jn], (short)0, accg[jm][jn], false,
            false);
        accd[jm][jn] = __builtin_amdgcn_wmma_f32_16x16x32_f16(
            false, af[jm], false, bdf[jn], (short)0, accd[jm][jn], false,
            false);
      }
    __syncthreads();
  }

  _Float16* He = hb + (size_t)e * T_ * DOUT_;
#pragma unroll
  for (int jm = 0; jm < 4; ++jm)
#pragma unroll
    for (int jn = 0; jn < 2; ++jn) {
      const int n = n0 + wn + jn * 16 + lmod;
#pragma unroll
      for (int i = 0; i < 8; ++i) {
        const int m = m0 + wm + jm * 16 + lhi * 8 + i;
        const float g = accg[jm][jn][i];
        const float u = accd[jm][jn][i];
        const float h = (g / (1.0f + __expf(-g))) * u;
        He[(size_t)m * DOUT_ + n] = (_Float16)h;
      }
    }
}

// ---------------------------------------------------------------------------
// GEMM2: out = h @ Wu, f32 out. Same tiling, single B matrix.
// ---------------------------------------------------------------------------
__global__ __launch_bounds__(256) void k_gemm2(const _Float16* __restrict__ hb,
                                               const _Float16* __restrict__ utb,
                                               float* __restrict__ out) {
  __shared__ __align__(16) _Float16 lds_a[2][128 * LSTR];
  __shared__ __align__(16) _Float16 lds_b[2][128 * LSTR];

  const int e = blockIdx.z;
  const int m0 = blockIdx.x * 128;
  const int n0 = blockIdx.y * 128;
  const _Float16* Ae = hb + (size_t)e * T_ * DOUT_;
  const _Float16* Be = utb + (size_t)e * DIN_ * DOUT_ + (size_t)n0 * DOUT_;

  const int tid = threadIdx.x;
  const int lane = tid & 31;
  const int w = tid >> 5;
  const int wm = (w & 1) * 64;
  const int wn = (w >> 1) * 32;
  const int lmod = lane & 15;
  const int lhi = lane >> 4;

  const int row0 = tid >> 2;
  const int row1 = (tid + 256) >> 2;
  const int col = (tid & 3) * 8;

  unsigned la[2], lb[2];
#pragma unroll
  for (int b = 0; b < 2; ++b) {
    la[b] = lds_base_u32(&lds_a[b][0]);
    lb[b] = lds_base_u32(&lds_b[b][0]);
  }

  auto stage = [&](int buf, int k0) {
    const unsigned o0 = (unsigned)(row0 * LSTR + col) * 2u;
    const unsigned o1 = (unsigned)(row1 * LSTR + col) * 2u;
    async_ld16(la[buf] + o0, Ae, (unsigned)((m0 + row0) * DOUT_ + k0 + col) * 2u);
    async_ld16(la[buf] + o1, Ae, (unsigned)((m0 + row1) * DOUT_ + k0 + col) * 2u);
    async_ld16(lb[buf] + o0, Be, (unsigned)(row0 * DOUT_ + k0 + col) * 2u);
    async_ld16(lb[buf] + o1, Be, (unsigned)(row1 * DOUT_ + k0 + col) * 2u);
  };

  v8f acc[4][2] = {};

  stage(0, 0);
  const int KT = DOUT_ / 32;
  for (int kt = 0; kt < KT; ++kt) {
    const int cur = kt & 1;
    if (kt + 1 < KT) {
      stage(cur ^ 1, (kt + 1) * 32);
      wait_async4();
    } else {
      wait_async0();
    }
    __syncthreads();

    const _Float16* A = &lds_a[cur][0];
    const _Float16* B = &lds_b[cur][0];

    v16h af[4], bf[2];
#pragma unroll
    for (int jm = 0; jm < 4; ++jm)
      af[jm] = load_a_frag(A, wm + jm * 16 + lmod, lane);
#pragma unroll
    for (int jn = 0; jn < 2; ++jn)
      bf[jn] = load_b_frag(B, wn + jn * 16 + lmod, lane);
#pragma unroll
    for (int jm = 0; jm < 4; ++jm)
#pragma unroll
      for (int jn = 0; jn < 2; ++jn)
        acc[jm][jn] = __builtin_amdgcn_wmma_f32_16x16x32_f16(
            false, af[jm], false, bf[jn], (short)0, acc[jm][jn], false, false);
    __syncthreads();
  }

  float* Oe = out + (size_t)e * T_ * DIN_;
#pragma unroll
  for (int jm = 0; jm < 4; ++jm)
#pragma unroll
    for (int jn = 0; jn < 2; ++jn) {
      const int n = n0 + wn + jn * 16 + lmod;
#pragma unroll
      for (int i = 0; i < 8; ++i) {
        const int m = m0 + wm + jm * 16 + lhi * 8 + i;
        Oe[(size_t)m * DIN_ + n] = acc[jm][jn][i];
      }
    }
}

// ---------------------------------------------------------------------------
extern "C" void kernel_launch(void* const* d_in, const int* in_sizes, int n_in,
                              void* d_out, int out_size, void* d_ws,
                              size_t ws_size, hipStream_t stream) {
  (void)in_sizes;
  (void)n_in;
  (void)out_size;
  (void)ws_size;

  const float* x = (const float*)d_in[0];
  const float* gate = (const float*)d_in[1];
  const float* down = (const float*)d_in[2];
  const float* up = (const float*)d_in[3];
  float* out = (float*)d_out;

  char* ws = (char*)d_ws;
  const size_t xb_sz = (size_t)E_ * T_ * DIN_ * sizeof(_Float16);
  const size_t w_sz = (size_t)E_ * DIN_ * DOUT_ * sizeof(_Float16);
  _Float16* xb = (_Float16*)(ws);
  _Float16* gtb = (_Float16*)(ws + xb_sz);
  _Float16* dtb = (_Float16*)(ws + xb_sz + w_sz);
  _Float16* utb = (_Float16*)(ws + xb_sz + 2 * w_sz);
  _Float16* hb = (_Float16*)(ws + xb_sz + 3 * w_sz);

  const long long nx = (long long)E_ * T_ * DIN_;
  k_cvt_f16<<<(unsigned)((nx / 4 + 255) / 256), 256, 0, stream>>>(x, xb, nx);

  dim3 tb(32, 8);
  k_transpose_cvt<<<dim3(DIN_ / 32, DOUT_ / 32, E_), tb, 0, stream>>>(
      gate, gtb, DIN_, DOUT_);
  k_transpose_cvt<<<dim3(DIN_ / 32, DOUT_ / 32, E_), tb, 0, stream>>>(
      down, dtb, DIN_, DOUT_);
  k_transpose_cvt<<<dim3(DOUT_ / 32, DIN_ / 32, E_), tb, 0, stream>>>(
      up, utb, DOUT_, DIN_);

  k_gemm1_swiglu<<<dim3(T_ / 128, DOUT_ / 128, E_), 256, 0, stream>>>(
      xb, gtb, dtb, hb);
  k_gemm2<<<dim3(T_ / 128, DIN_ / 128, E_), 256, 0, stream>>>(hb, utb, out);
}
